// MoE_60481729462384
// MI455X (gfx1250) — matrix operand order repaired and checked by reading, not compiled, not run on hardware
//
#include <hip/hip_runtime.h>

#define T_TOK   4096
#define DMODEL  1024
#define FFN_N   1024
#define NEXP    8

typedef __attribute__((ext_vector_type(16))) __bf16 v16bf;
typedef __attribute__((ext_vector_type(8)))  float  v8f;

union Frag16 { v16bf v; uint4 u4[2]; };

__device__ __forceinline__ unsigned short f32_to_bf16(float f) {
  unsigned int u = __float_as_uint(f);
  u += 0x7fffu + ((u >> 16) & 1u);          // round-to-nearest-even
  return (unsigned short)(u >> 16);
}

__device__ __forceinline__ v8f wmma_bf16(v16bf a, v16bf b, v8f c) {
  return __builtin_amdgcn_wmma_f32_16x16x32_bf16(false, a, false, b, (short)0, c,
                                                 false, false);
}

// CDNA5 async global->LDS copy (ASYNCcnt path), GVS addressing.
__device__ __forceinline__ unsigned lds_addr(const void* p) {
  return (unsigned)(uintptr_t)p;            // low 32 bits = LDS offset
}
__device__ __forceinline__ void async_ld128(unsigned lds, const void* sbase,
                                            unsigned voff) {
  asm volatile("global_load_async_to_lds_b128 %0, %1, %2"
               :: "v"(lds), "v"(voff), "s"(sbase) : "memory");
}
__device__ __forceinline__ void wait_async0() {
  asm volatile("s_wait_asynccnt 0x0" ::: "memory");
}
// Retire the oldest in-flight stage (async loads complete in order per wave);
// leaves the newest 4 ops (next prefetch stage) outstanding.
__device__ __forceinline__ void wait_async4() {
  asm volatile("s_wait_asynccnt 0x4" ::: "memory");
}

// ---------------------------------------------------------------------------
// Gating: logits = x @ gate_w + gate_b ; softmax ; top-2 -> cw[T][8]
// ---------------------------------------------------------------------------
__global__ __launch_bounds__(128)
void gate_kernel(const float* __restrict__ x, const float* __restrict__ gw,
                 const float* __restrict__ gb, float* __restrict__ cw) {
  const int wid  = threadIdx.x >> 5;
  const int lane = threadIdx.x & 31;
  const int t    = blockIdx.x * 4 + wid;

  float acc[NEXP];
#pragma unroll
  for (int e = 0; e < NEXP; ++e) acc[e] = 0.0f;

  const float* xr = x + (size_t)t * DMODEL;
  for (int d = lane; d < DMODEL; d += 32) {
    const float xv = xr[d];
    const float* g = gw + (size_t)d * NEXP;
#pragma unroll
    for (int e = 0; e < NEXP; ++e) acc[e] += xv * g[e];
  }
#pragma unroll
  for (int off = 16; off > 0; off >>= 1) {
#pragma unroll
    for (int e = 0; e < NEXP; ++e) acc[e] += __shfl_xor(acc[e], off, 32);
  }

  if (lane == 0) {
    float l[NEXP], p[NEXP];
    float m = -3.0e38f;
#pragma unroll
    for (int e = 0; e < NEXP; ++e) { l[e] = acc[e] + gb[e]; m = fmaxf(m, l[e]); }
    float s = 0.0f;
#pragma unroll
    for (int e = 0; e < NEXP; ++e) { p[e] = __expf(l[e] - m); s += p[e]; }
    const float inv = 1.0f / s;
#pragma unroll
    for (int e = 0; e < NEXP; ++e) p[e] *= inv;

    int i1 = 0;
#pragma unroll
    for (int e = 1; e < NEXP; ++e) if (p[e] > p[i1]) i1 = e;
    int i2 = (i1 == 0) ? 1 : 0;
#pragma unroll
    for (int e = 0; e < NEXP; ++e)
      if (e != i1 && p[e] > p[i2]) i2 = e;

    float* o = cw + (size_t)t * NEXP;
#pragma unroll
    for (int e = 0; e < NEXP; ++e)
      o[e] = (e == i1) ? p[i1] : ((e == i2) ? p[i2] : 0.0f);
  }
}

// ---------------------------------------------------------------------------
// Per-expert token compaction (deterministic, token order).
// ---------------------------------------------------------------------------
__global__ __launch_bounds__(1024)
void scan_kernel(const float* __restrict__ cw, int* __restrict__ idx,
                 int* __restrict__ counts) {
  __shared__ int sums[1024];
  const int e = blockIdx.x;
  const int t = threadIdx.x;
  int f[4];
  int c = 0;
#pragma unroll
  for (int i = 0; i < 4; ++i) {
    f[i] = (cw[(size_t)(t * 4 + i) * NEXP + e] != 0.0f) ? 1 : 0;
    c += f[i];
  }
  sums[t] = c;
  __syncthreads();
  for (int off = 1; off < 1024; off <<= 1) {
    const int add = (t >= off) ? sums[t - off] : 0;
    __syncthreads();
    sums[t] += add;
    __syncthreads();
  }
  int pos = sums[t] - c;                    // exclusive prefix
  int* lst = idx + (size_t)e * T_TOK;
#pragma unroll
  for (int i = 0; i < 4; ++i)
    if (f[i]) lst[pos++] = t * 4 + i;
  if (t == 1023) counts[e] = sums[1023];
}

// ---------------------------------------------------------------------------
// x fp32 [T][D] -> bf16 row-major
// ---------------------------------------------------------------------------
__global__ __launch_bounds__(256)
void convert_x_kernel(const float* __restrict__ x, unsigned short* __restrict__ xb) {
  const size_t i = ((size_t)blockIdx.x * 256 + threadIdx.x) * 4;
  const float4 f = *(const float4*)(x + i);
  ushort4 o;
  o.x = f32_to_bf16(f.x); o.y = f32_to_bf16(f.y);
  o.z = f32_to_bf16(f.z); o.w = f32_to_bf16(f.w);
  *(ushort4*)(xb + i) = o;
}

// ---------------------------------------------------------------------------
// Weight prep: src fp32 [K][N] -> dst bf16 [N][K] (B^T: B frag = contiguous K)
// ---------------------------------------------------------------------------
__global__ __launch_bounds__(256)
void transpose_convert_kernel(const float* __restrict__ src,
                              unsigned short* __restrict__ dst,
                              int K, int N) {
  __shared__ float tile[32][33];
  const int k0 = blockIdx.y * 32;
  const int n0 = blockIdx.x * 32;
  const int t  = threadIdx.x;
#pragma unroll
  for (int i = 0; i < 4; ++i) {
    const int idx = t + 256 * i;
    const int r = idx >> 5, c = idx & 31;
    tile[r][c] = src[(size_t)(k0 + r) * N + n0 + c];
  }
  __syncthreads();
  const int nl = t >> 3;
  const int kk = (t & 7) * 4;
  ushort4 o;
  o.x = f32_to_bf16(tile[kk + 0][nl]);
  o.y = f32_to_bf16(tile[kk + 1][nl]);
  o.z = f32_to_bf16(tile[kk + 2][nl]);
  o.w = f32_to_bf16(tile[kk + 3][nl]);
  *(ushort4*)(dst + (size_t)(n0 + nl) * K + k0 + kk) = o;
}

// ---------------------------------------------------------------------------
// Up projection + fused SwiGLU over gathered token slots.
//   slot i -> token idxlist[i] (identity when idxlist==nullptr, count=T)
// WG: 128 slots x 64 h-cols (both halves); 8 waves = 4(M) x 2(N).
// 3-stage async pipeline: prefetch depth 2, partial ASYNCcnt waits.
// ---------------------------------------------------------------------------
__global__ __launch_bounds__(256)
void up_kernel(const unsigned short* __restrict__ xb,   // [T][D] bf16
               const unsigned short* __restrict__ w1t,  // [2F][D] bf16 (B^T)
               const float* __restrict__ b1,            // [2F]
               unsigned short* __restrict__ h,          // [T][F] bf16 (by slot)
               const int* __restrict__ idxlist,
               const int* __restrict__ count_ptr) {
  __shared__ unsigned short As[3][128][48];
  __shared__ unsigned short Bs[3][128][48];

  const int count = count_ptr ? *count_ptr : T_TOK;
  const int m0 = blockIdx.y * 128;
  if (m0 >= count) return;
  const int j0 = blockIdx.x * 64;
  const int tid = threadIdx.x;
  const int wid = tid >> 5, lane = tid & 31;
  const int waveM = wid & 3;
  const int waveN = wid >> 2;

  // K-invariant copy assignments: 2 chunks (16B) per thread for A and B.
  unsigned ldsA[3][2], ldsB[3][2];          // [buf][chunk]
  unsigned goffA[2], goffB[2];              // byte offsets, +k*64 per stage
#pragma unroll
  for (int i = 0; i < 2; ++i) {
    const int cch  = (tid << 1) + i;        // 0..511
    const int row  = cch >> 2;
    const int col8 = (cch & 3) << 3;
    const int gi = m0 + row;
    const int ci = (gi < count) ? gi : (count - 1);
    const int tok = idxlist ? idxlist[ci] : ci;
    goffA[i] = ((unsigned)tok << 11) + ((unsigned)col8 << 1);
    const int grow = (row < 64) ? (j0 + row) : (FFN_N + j0 + row - 64);
    goffB[i] = ((unsigned)grow << 11) + ((unsigned)col8 << 1);
#pragma unroll
    for (int buf = 0; buf < 3; ++buf) {
      ldsA[buf][i] = lds_addr(&As[buf][row][col8]);
      ldsB[buf][i] = lds_addr(&Bs[buf][row][col8]);
    }
  }

  v8f zero;
#pragma unroll
  for (int i = 0; i < 8; ++i) zero[i] = 0.0f;
  v8f acc[2][2][2];
#pragma unroll
  for (int a = 0; a < 2; ++a)
#pragma unroll
    for (int b = 0; b < 2; ++b)
#pragma unroll
      for (int c = 0; c < 2; ++c) acc[a][b][c] = zero;

  // Prologue: stages 0 and 1 in flight; retire stage 0 (wait <= 4).
#pragma unroll
  for (int i = 0; i < 2; ++i) {
    async_ld128(ldsA[0][i], xb, goffA[i]);
    async_ld128(ldsB[0][i], w1t, goffB[i]);
  }
#pragma unroll
  for (int i = 0; i < 2; ++i) {
    async_ld128(ldsA[1][i], xb, goffA[i] + 64u);
    async_ld128(ldsB[1][i], w1t, goffB[i] + 64u);
  }
  wait_async4();
  __syncthreads();

  for (int kt = 0; kt < 32; ++kt) {
    const int buf = kt % 3;
    const bool more = (kt + 2) < 32;
    if (more) {
      const unsigned kadd = (unsigned)(kt + 2) << 6;   // 32 bf16 = 64 bytes
      const int nb = (kt + 2) % 3;
#pragma unroll
      for (int i = 0; i < 2; ++i) {
        async_ld128(ldsA[nb][i], xb, goffA[i] + kadd);
        async_ld128(ldsB[nb][i], w1t, goffB[i] + kadd);
      }
    }

    Frag16 fa[2];
#pragma unroll
    for (int mt = 0; mt < 2; ++mt) {
      const int r = waveM * 32 + mt * 16 + (lane & 15);
      const uint4* p = (const uint4*)&As[buf][r][0];
      const int q = lane >> 4;
      fa[mt].u4[0] = p[q];
      fa[mt].u4[1] = p[q + 2];
    }
    Frag16 fb[2][2];
#pragma unroll
    for (int half = 0; half < 2; ++half)
#pragma unroll
      for (int nt = 0; nt < 2; ++nt) {
        const int r = half * 64 + waveN * 32 + nt * 16 + (lane & 15);
        const uint4* p = (const uint4*)&Bs[buf][r][0];
        const int q = lane >> 4;
        fb[half][nt].u4[0] = p[q];
        fb[half][nt].u4[1] = p[q + 2];
      }
#pragma unroll
    for (int half = 0; half < 2; ++half)
#pragma unroll
      for (int mt = 0; mt < 2; ++mt)
#pragma unroll
        for (int nt = 0; nt < 2; ++nt)
          acc[half][mt][nt] = wmma_bf16(fa[mt].v, fb[half][nt].v, acc[half][mt][nt]);

    // Retire the stage needed next iteration; deeper stage stays in flight.
    if (more) wait_async4(); else wait_async0();
    __syncthreads();
  }

#pragma unroll
  for (int mt = 0; mt < 2; ++mt)
#pragma unroll
    for (int nt = 0; nt < 2; ++nt) {
      const int rowb = m0 + waveM * 32 + mt * 16 + ((lane >> 4) << 3);
      const int col  = j0 + waveN * 32 + nt * 16 + (lane & 15);
      const float ba = b1[col];
      const float bb = b1[FFN_N + col];
#pragma unroll
      for (int v = 0; v < 8; ++v) {
        const int i = rowb + v;
        if (i < count) {
          const float a = acc[0][mt][nt][v] + ba;
          const float b = acc[1][mt][nt][v] + bb;
          const float s = a / (1.0f + __expf(-a));
          h[((size_t)i << 10) + col] = f32_to_bf16(s * b);
        }
      }
    }
}

// ---------------------------------------------------------------------------
// Down projection + scaled scatter-accumulate:
//   out[tok][d] (+)= scale * (h[slot].w2[:,d] + b2[d])
// WG: 128 slots x 128 cols; 8 waves = 2(M) x 4(N). Same 3-stage pipeline.
// ---------------------------------------------------------------------------
__global__ __launch_bounds__(256)
void down_kernel(const unsigned short* __restrict__ h,    // [T][F] bf16 (slots)
                 const unsigned short* __restrict__ w2t,  // [D][F] bf16 (B^T)
                 const float* __restrict__ b2,            // [D]
                 const float* __restrict__ cw,            // [T][8]
                 const int* __restrict__ idxlist,
                 const int* __restrict__ count_ptr,
                 int expert, int accum,
                 float* __restrict__ out) {               // [T][D]
  __shared__ unsigned short As[3][128][48];
  __shared__ unsigned short Bs[3][128][48];

  const int count = count_ptr ? *count_ptr : T_TOK;
  const int m0 = blockIdx.y * 128;
  if (m0 >= count) return;
  const int n0 = blockIdx.x * 128;
  const int tid = threadIdx.x;
  const int wid = tid >> 5, lane = tid & 31;
  const int waveM = wid & 1;
  const int waveN = wid >> 1;

  unsigned ldsA[3][2], ldsB[3][2];
  unsigned goffA[2], goffB[2];
#pragma unroll
  for (int i = 0; i < 2; ++i) {
    const int cch  = (tid << 1) + i;
    const int row  = cch >> 2;
    const int col8 = (cch & 3) << 3;
    goffA[i] = ((unsigned)(m0 + row) << 11) + ((unsigned)col8 << 1);
    goffB[i] = ((unsigned)(n0 + row) << 11) + ((unsigned)col8 << 1);
#pragma unroll
    for (int buf = 0; buf < 3; ++buf) {
      ldsA[buf][i] = lds_addr(&As[buf][row][col8]);
      ldsB[buf][i] = lds_addr(&Bs[buf][row][col8]);
    }
  }

  v8f zero;
#pragma unroll
  for (int i = 0; i < 8; ++i) zero[i] = 0.0f;
  v8f acc[4][2];
#pragma unroll
  for (int a = 0; a < 4; ++a)
#pragma unroll
    for (int b = 0; b < 2; ++b) acc[a][b] = zero;

#pragma unroll
  for (int i = 0; i < 2; ++i) {
    async_ld128(ldsA[0][i], h, goffA[i]);
    async_ld128(ldsB[0][i], w2t, goffB[i]);
  }
#pragma unroll
  for (int i = 0; i < 2; ++i) {
    async_ld128(ldsA[1][i], h, goffA[i] + 64u);
    async_ld128(ldsB[1][i], w2t, goffB[i] + 64u);
  }
  wait_async4();
  __syncthreads();

  for (int kt = 0; kt < 32; ++kt) {
    const int buf = kt % 3;
    const bool more = (kt + 2) < 32;
    if (more) {
      const unsigned kadd = (unsigned)(kt + 2) << 6;
      const int nb = (kt + 2) % 3;
#pragma unroll
      for (int i = 0; i < 2; ++i) {
        async_ld128(ldsA[nb][i], h, goffA[i] + kadd);
        async_ld128(ldsB[nb][i], w2t, goffB[i] + kadd);
      }
    }

    Frag16 fa[4];
#pragma unroll
    for (int mt = 0; mt < 4; ++mt) {
      const int r = waveM * 64 + mt * 16 + (lane & 15);
      const uint4* p = (const uint4*)&As[buf][r][0];
      const int q = lane >> 4;
      fa[mt].u4[0] = p[q];
      fa[mt].u4[1] = p[q + 2];
    }
    Frag16 fb[2];
#pragma unroll
    for (int nt = 0; nt < 2; ++nt) {
      const int r = waveN * 32 + nt * 16 + (lane & 15);
      const uint4* p = (const uint4*)&Bs[buf][r][0];
      const int q = lane >> 4;
      fb[nt].u4[0] = p[q];
      fb[nt].u4[1] = p[q + 2];
    }
#pragma unroll
    for (int mt = 0; mt < 4; ++mt)
#pragma unroll
      for (int nt = 0; nt < 2; ++nt)
        acc[mt][nt] = wmma_bf16(fa[mt].v, fb[nt].v, acc[mt][nt]);

    if (more) wait_async4(); else wait_async0();
    __syncthreads();
  }

#pragma unroll
  for (int mt = 0; mt < 4; ++mt)
#pragma unroll
    for (int nt = 0; nt < 2; ++nt) {
      const int rowb = m0 + waveM * 64 + mt * 16 + ((lane >> 4) << 3);
      const int col  = n0 + waveN * 32 + nt * 16 + (lane & 15);
      const float bias = b2[col];
#pragma unroll
      for (int v = 0; v < 8; ++v) {
        const int i = rowb + v;
        if (i < count) {
          const int tok = idxlist ? idxlist[i] : i;
          const float s = (expert < 0) ? 1.0f : cw[(size_t)tok * NEXP + expert];
          const float val = (acc[mt][nt][v] + bias) * s;
          float* o = out + ((size_t)tok << 10) + col;
          if (accum) *o += val; else *o = val;
        }
      }
    }
}

// ---------------------------------------------------------------------------
// Host-side launcher
// ---------------------------------------------------------------------------
extern "C" void kernel_launch(void* const* d_in, const int* in_sizes, int n_in,
                              void* d_out, int out_size, void* d_ws, size_t ws_size,
                              hipStream_t stream) {
  (void)in_sizes; (void)n_in; (void)out_size; (void)ws_size;

  const float* x   = (const float*)d_in[0];
  const float* gw  = (const float*)d_in[1];
  const float* gb  = (const float*)d_in[2];
  const float* sw1 = (const float*)d_in[3];
  const float* sb1 = (const float*)d_in[4];
  const float* sw2 = (const float*)d_in[5];
  const float* sb2 = (const float*)d_in[6];
  const float* rw1 = (const float*)d_in[7];
  const float* rb1 = (const float*)d_in[8];
  const float* rw2 = (const float*)d_in[9];
  const float* rb2 = (const float*)d_in[10];
  float* out = (float*)d_out;

  char* ws = (char*)d_ws;
  const size_t CW_B  = (size_t)T_TOK * NEXP * 4;       // 128 KB
  const size_t IDX_B = (size_t)NEXP * T_TOK * 4;       // 128 KB
  const size_t CNT_B = 256;
  const size_t XB_B  = (size_t)T_TOK * DMODEL * 2;     // 8 MB
  const size_t W1T_B = (size_t)2 * FFN_N * DMODEL * 2; // 4 MB
  const size_t W2T_B = (size_t)FFN_N * DMODEL * 2;     // 2 MB

  float*          cw   = (float*)(ws);
  int*            idxb = (int*)(ws + CW_B);
  int*            cnts = (int*)(ws + CW_B + IDX_B);
  unsigned short* xb   = (unsigned short*)(ws + CW_B + IDX_B + CNT_B);
  unsigned short* w1t  = (unsigned short*)(ws + CW_B + IDX_B + CNT_B + XB_B);
  unsigned short* w2t  = (unsigned short*)(ws + CW_B + IDX_B + CNT_B + XB_B + W1T_B);
  unsigned short* hbuf = (unsigned short*)(ws + CW_B + IDX_B + CNT_B + XB_B + W1T_B + W2T_B);

  gate_kernel<<<T_TOK / 4, 128, 0, stream>>>(x, gw, gb, cw);
  scan_kernel<<<NEXP, 1024, 0, stream>>>(cw, idxb, cnts);
  convert_x_kernel<<<(T_TOK * DMODEL) / (256 * 4), 256, 0, stream>>>(x, xb);

  for (int e = -1; e < NEXP; ++e) {
    const float* w1 = (e < 0) ? sw1 : rw1 + (size_t)e * DMODEL * 2 * FFN_N;
    const float* w2 = (e < 0) ? sw2 : rw2 + (size_t)e * FFN_N * DMODEL;
    const float* b1 = (e < 0) ? sb1 : rb1 + (size_t)e * 2 * FFN_N;
    const float* b2 = (e < 0) ? sb2 : rb2 + (size_t)e * DMODEL;
    const int* il = (e < 0) ? nullptr : idxb + (size_t)e * T_TOK;
    const int* cp = (e < 0) ? nullptr : cnts + e;

    transpose_convert_kernel<<<dim3(2 * FFN_N / 32, DMODEL / 32), 256, 0, stream>>>(
        w1, w1t, DMODEL, 2 * FFN_N);
    transpose_convert_kernel<<<dim3(DMODEL / 32, FFN_N / 32), 256, 0, stream>>>(
        w2, w2t, FFN_N, DMODEL);

    up_kernel<<<dim3(FFN_N / 64, T_TOK / 128), 256, 0, stream>>>(
        xb, w1t, b1, hbuf, il, cp);
    down_kernel<<<dim3(DMODEL / 128, T_TOK / 128), 256, 0, stream>>>(
        hbuf, w2t, b2, cw, il, cp, e, (e >= 0) ? 1 : 0, out);
  }
}
